// RVAE_21827023798714
// MI455X (gfx1250) — matrix operand order repaired and checked by reading, not compiled, not run on hardware
//
#include <hip/hip_runtime.h>
#include <hip/hip_bf16.h>
#include <math.h>

typedef __attribute__((ext_vector_type(16))) __bf16 v16bf;
typedef __attribute__((ext_vector_type(8)))  float  v8f;
typedef unsigned int u32x4 __attribute__((ext_vector_type(4)));
typedef int          i32x8 __attribute__((ext_vector_type(8)));
typedef int          i32x4 __attribute__((ext_vector_type(4)));

#define T_STEPS 512
#define B_TOT   256
#define F_DIM   128
#define R_DIM   256
#define L_DIM   64

// ---- bf16 weight arena offsets in d_ws (ushort elements) ----
static constexpr size_t OFF_PXW1 = 0;          // 256x128
static constexpr size_t OFF_PXW2 = 32768;      // 256x256
static constexpr size_t OFF_PZW1 = 98304;      // 256x64
static constexpr size_t OFF_PZW2 = 114688;     // 256x256
static constexpr size_t OFF_EW1  = 180224;     // 256x512
static constexpr size_t OFF_EW2  = 311296;     // 256x256
static constexpr size_t OFF_EMW  = 376832;     // 64x256
static constexpr size_t OFF_ESW  = 393216;     // 64x256
static constexpr size_t OFF_PRW  = 409600;     // 256x256
static constexpr size_t OFF_PMW  = 475136;     // 64x256
static constexpr size_t OFF_PSW  = 491520;     // 64x256
static constexpr size_t OFF_DW1  = 507904;     // 256x512
static constexpr size_t OFF_DW2  = 638976;     // 256x256
static constexpr size_t OFF_DMW  = 704512;     // 128x256
static constexpr size_t OFF_DSW  = 737280;     // 128x256
static constexpr size_t OFF_GIH0 = 770048;     // 768x512
static constexpr size_t OFF_GHH0 = 1163264;    // 768x256
static constexpr size_t OFF_GIH1 = 1359872;    // 768x256
static constexpr size_t OFF_GHH1 = 1556480;    // 768x256

static constexpr size_t TBL = (size_t)T_STEPS * B_TOT * L_DIM;   // em/es elems
static constexpr size_t TBF = (size_t)T_STEPS * B_TOT * F_DIM;   // dm/ds elems

__device__ __forceinline__ unsigned short f2bf(float f) {
  unsigned u = __float_as_uint(f);
  u += 0x7FFFu + ((u >> 16) & 1u);          // round-to-nearest-even
  return (unsigned short)(u >> 16);
}
__device__ __forceinline__ float bf2f(unsigned short h) {
  return __uint_as_float(((unsigned)h) << 16);
}
__device__ __forceinline__ float sigmf(float x) { return 1.f / (1.f + expf(-x)); }
__device__ __forceinline__ float softplusf(float x) { return (x > 20.f) ? x : log1pf(expf(x)); }

// TDM: async DMA of a contiguous 16x64 fp32 tile (row stride 64 elems) into LDS.
// D# per CDNA5 ISA ch.8: group0 = {count=1 | lds_addr | global_addr | type=2},
// group1 = {data_size=4B, tensor 64x16, tile 64x16, dim0_stride=64}.
// This toolchain exposes the 6-arg builtin form:
//   (uint32x4 g0, int32x8 g1, int32x4 g2, int32x4 g3, int32x8 extra, i32 cpol)
__device__ __forceinline__ void tdm_load_16x64_f32(const float* gsrc, unsigned lds_byte_off) {
  unsigned long long ga = (unsigned long long)(uintptr_t)gsrc;
  u32x4 g0;
  g0[0] = 1u;                                        // count=1, user mode
  g0[1] = lds_byte_off;                              // lds_addr (bytes)
  g0[2] = (unsigned)(ga & 0xFFFFFFFFu);              // global_addr[31:0]
  g0[3] = (unsigned)((ga >> 32) & 0x01FFFFFFu)       // global_addr[56:32]
          | 0x80000000u;                             // type=2 ("image")
  i32x8 g1;
  g1[0] = 0x00020000;                                // data_size=2 (4B), no mask/pad
  g1[1] = (int)(64u << 16);                          // tensor_dim0[15:0] @ bits63:48
  g1[2] = (int)(16u << 16);                          // tensor_dim0 hi=0 | tensor_dim1 lo @95:80
  g1[3] = (int)(64u << 16);                          // tensor_dim1 hi=0 | tile_dim0 @127:112
  g1[4] = 16;                                        // tile_dim1 @143:128, tile_dim2=0
  g1[5] = 64;                                        // tensor_dim0_stride lo32
  g1[6] = 0;                                         // stride hi16 | dim1_stride lo
  g1[7] = 0;
  i32x4 gz4 = {0, 0, 0, 0};
  i32x8 gz8;
  gz8[0] = 0; gz8[1] = 0; gz8[2] = 0; gz8[3] = 0;
  gz8[4] = 0; gz8[5] = 0; gz8[6] = 0; gz8[7] = 0;
  __builtin_amdgcn_tensor_load_to_lds(g0, g1, gz4, gz4, gz8, 0);
}

// A chunk mapper: logical k -> physical address across a 2-segment concat.
__device__ __forceinline__ const unsigned short* amap(const unsigned short* A0, int ld0,
                                                      const unsigned short* A1, int ld1,
                                                      int split, int m, int k) {
  return (k < split) ? (A0 + m * ld0 + k) : (A1 + m * ld1 + (k - split));
}

// One 16x16 output tile: D += A(16xK) * W^T tile.  CDNA5 wave32 layouts:
//  A: lane m=lane&15; chunks (8 halfs) at k0=kb+((lane>>4)<<3) and k0+16
//  B: lane n=nbase+(lane&15); chunks (8 halfs) at k0=kb+((lane>>4)<<3) and k0+16
__device__ __forceinline__ v8f wmma_tile(const unsigned short* A0, int ld0,
                                         const unsigned short* A1, int ld1, int asplit,
                                         const unsigned short* W0, int ldw0,
                                         const unsigned short* W1, int ldw1, int wsplit,
                                         int K, int nbase, int lane, v8f acc) {
  const int m     = lane & 15;
  const int nrow  = nbase + (lane & 15);
  const int ahalf = (lane >> 4) << 3;
  const int bhalf = (lane >> 4) << 3;
  for (int kb = 0; kb < K; kb += 32) {
    union { uint4 u[2]; v16bf v; } a, b;
    a.u[0] = *(const uint4*)amap(A0, ld0, A1, ld1, asplit, m, kb + ahalf);
    a.u[1] = *(const uint4*)amap(A0, ld0, A1, ld1, asplit, m, kb + ahalf + 16);
    int kw = kb + bhalf;
    const unsigned short* wp = (kw < wsplit) ? (W0 + (size_t)nrow * ldw0 + kw)
                                             : (W1 + (size_t)nrow * ldw1 + (kw - wsplit));
    b.u[0] = *(const uint4*)wp;
    b.u[1] = *(const uint4*)(wp + 16);
    acc = __builtin_amdgcn_wmma_f32_16x16x32_bf16(false, a.v, false, b.v,
                                                  (short)0, acc, false, false);
  }
  return acc;
}

// C layout: lane n=lane&15, vgpr r -> row m = r + ((lane>>4)<<3)
__device__ __forceinline__ void relu_store(unsigned short* dst, int ldd, int col0,
                                           int tile, int lane, const float* bias, v8f acc) {
  const int nloc = tile * 16 + (lane & 15);
  const float b  = bias[nloc];
  const int n    = col0 + nloc;
  const int mb   = (lane >> 4) << 3;
#pragma unroll
  for (int r = 0; r < 8; ++r)
    dst[(mb + r) * ldd + n] = f2bf(fmaxf(acc[r] + b, 0.f));
}

__global__ void __launch_bounds__(256)
vrnn_persistent_kernel(const float* __restrict__ xg, const float* __restrict__ epsg,
                       const unsigned short* __restrict__ wb,
                       const float* __restrict__ pxb1, const float* __restrict__ pxb2,
                       const float* __restrict__ pzb1, const float* __restrict__ pzb2,
                       const float* __restrict__ eb1,  const float* __restrict__ eb2,
                       const float* __restrict__ emb,  const float* __restrict__ esb,
                       const float* __restrict__ prb,  const float* __restrict__ pmb,
                       const float* __restrict__ psb,  const float* __restrict__ db1,
                       const float* __restrict__ db2,  const float* __restrict__ dmb,
                       const float* __restrict__ dsb,  float* __restrict__ out) {
  // LDS (bf16 as ushort + fp32 eps staging): 63,488 bytes total
  __shared__ __align__(16) unsigned short sAg[16 * 1024]; // [phi_x|phi_z|h0|h1]
  __shared__ __align__(16) unsigned short sT1[16 * 256];
  __shared__ __align__(16) unsigned short sT2[16 * 256];
  __shared__ __align__(16) unsigned short sX [16 * 128];
  __shared__ __align__(16) unsigned short sZ [16 * 64];
  __shared__ __align__(16) float          sEps[2][16 * 64]; // TDM double buffer

  const int tid  = threadIdx.x;
  const int lane = tid & 31;
  const int wave = tid >> 5;
  const int b0   = blockIdx.x * 16;

  const unsigned short* wPXW1 = wb + OFF_PXW1;
  const unsigned short* wPXW2 = wb + OFF_PXW2;
  const unsigned short* wPZW1 = wb + OFF_PZW1;
  const unsigned short* wPZW2 = wb + OFF_PZW2;
  const unsigned short* wEW1  = wb + OFF_EW1;
  const unsigned short* wEW2  = wb + OFF_EW2;
  const unsigned short* wEMW  = wb + OFF_EMW;
  const unsigned short* wESW  = wb + OFF_ESW;
  const unsigned short* wPRW  = wb + OFF_PRW;
  const unsigned short* wPMW  = wb + OFF_PMW;
  const unsigned short* wPSW  = wb + OFF_PSW;
  const unsigned short* wDW1  = wb + OFF_DW1;
  const unsigned short* wDW2  = wb + OFF_DW2;
  const unsigned short* wDMW  = wb + OFF_DMW;
  const unsigned short* wDSW  = wb + OFF_DSW;
  const unsigned short* wGIH0 = wb + OFF_GIH0;
  const unsigned short* wGHH0 = wb + OFF_GHH0;
  const unsigned short* wGIH1 = wb + OFF_GIH1;
  const unsigned short* wGHH1 = wb + OFF_GHH1;

  float* em_out = out + 2;
  float* es_out = out + 2 + TBL;
  float* dm_out = out + 2 + 2 * TBL;
  float* ds_out = out + 2 + 2 * TBL + TBF;

  // h0,h1 = 0 (cols 512..1023 of sAg)
  for (int i = tid; i < 16 * 512; i += 256) {
    int m = i >> 9, c = i & 511;
    sAg[m * 1024 + 512 + c] = 0;
  }

  // Prologue: async-stage eps(t=0) into buffer 0 via the Tensor Data Mover.
  if (wave == 0)
    tdm_load_16x64_f32(epsg + (size_t)b0 * L_DIM,
                       (unsigned)(uintptr_t)&sEps[0][0]);
  __syncthreads();

  float kld_acc = 0.f, nll_acc = 0.f;
  const v8f vz = {};

#pragma unroll 1
  for (int t = 0; t < T_STEPS; ++t) {
    // S0: stage x_t -> LDS bf16; kick off TDM for eps(t+1); prefetch x(t+1)
    if (wave == 0 && t + 1 < T_STEPS)
      tdm_load_16x64_f32(epsg + ((size_t)((t + 1) * B_TOT) + b0) * L_DIM,
                         (unsigned)(uintptr_t)&sEps[(t + 1) & 1][0]);
    for (int i = tid; i < 16 * F_DIM; i += 256) {
      int m = i / F_DIM, f = i % F_DIM;
      sX[m * F_DIM + f] = f2bf(xg[((size_t)(t * B_TOT) + b0 + m) * F_DIM + f]);
    }
    if (t + 1 < T_STEPS && tid < 16)
      __builtin_prefetch(&xg[((size_t)((t + 1) * B_TOT) + b0 + tid) * F_DIM], 0, 3);
    __syncthreads();

    // S1: phi_x layer1 -> sT1
    for (int tl = wave; tl < 16; tl += 8)
      relu_store(sT1, 256, 0, tl, lane, pxb1,
                 wmma_tile(sX, 128, sX, 128, 128, wPXW1, 128, wPXW1, 128, 128, 128, tl * 16, lane, vz));
    __syncthreads();

    // S2: phi_x layer2 -> sAg[:,0:256]; prior hidden relu(h1 @ prw^T) -> sT2
    for (int tt = wave; tt < 32; tt += 8) {
      if (tt < 16) {
        relu_store(sAg, 1024, 0, tt, lane, pxb2,
                   wmma_tile(sT1, 256, sT1, 256, 256, wPXW2, 256, wPXW2, 256, 256, 256, tt * 16, lane, vz));
      } else {
        int tl = tt - 16;
        relu_store(sT2, 256, 0, tl, lane, prb,
                   wmma_tile(sAg + 768, 1024, sAg + 768, 1024, 256, wPRW, 256, wPRW, 256, 256, 256, tl * 16, lane, vz));
      }
    }
    __syncthreads();

    // S3a: prior heads pm/ps from sT2 (waves 0-3, kept in registers)
    v8f pmv = vz, psv = vz;
    if (wave < 4) {
      int nb = wave * 16;
      v8f a1 = wmma_tile(sT2, 256, sT2, 256, 256, wPMW, 256, wPMW, 256, 256, 256, nb, lane, vz);
      v8f a2 = wmma_tile(sT2, 256, sT2, 256, 256, wPSW, 256, wPSW, 256, 256, 256, nb, lane, vz);
      float bm = pmb[nb + (lane & 15)], bs = psb[nb + (lane & 15)];
#pragma unroll
      for (int r = 0; r < 8; ++r) { pmv[r] = sigmf(a1[r] + bm); psv[r] = softplusf(a2[r] + bs); }
    }
    __syncthreads();

    // S3b: enc layer1 over [phi_x|h1] (split-A K=512) -> sT2 (overwrites pr)
    for (int tl = wave; tl < 16; tl += 8)
      relu_store(sT2, 256, 0, tl, lane, eb1,
                 wmma_tile(sAg, 1024, sAg + 768, 1024, 256, wEW1, 512, wEW1, 512, 512, 512, tl * 16, lane, vz));
    __syncthreads();

    // S4: enc layer2 -> sT1 (eo); wave0 retires the eps(t) TDM transfer
    for (int tl = wave; tl < 16; tl += 8)
      relu_store(sT1, 256, 0, tl, lane, eb2,
                 wmma_tile(sT2, 256, sT2, 256, 256, wEW2, 256, wEW2, 256, 256, 256, tl * 16, lane, vz));
    if (wave == 0) {
      if (t + 1 < T_STEPS) __builtin_amdgcn_s_wait_tensorcnt(1); // eps(t) done, t+1 in flight
      else                 __builtin_amdgcn_s_wait_tensorcnt(0);
    }
    __syncthreads();

    // S5: enc heads + z sample + KLD (waves 0-3); eps comes from LDS (TDM)
    if (wave < 4) {
      int nb = wave * 16;
      v8f a1 = wmma_tile(sT1, 256, sT1, 256, 256, wEMW, 256, wEMW, 256, 256, 256, nb, lane, vz);
      v8f a2 = wmma_tile(sT1, 256, sT1, 256, 256, wESW, 256, wESW, 256, 256, 256, nb, lane, vz);
      int n = lane & 15, mb = (lane >> 4) << 3;
      float bm = emb[nb + n], bs = esb[nb + n];
      const float* epsl = &sEps[t & 1][0];
#pragma unroll
      for (int r = 0; r < 8; ++r) {
        int m = mb + r;
        size_t gi = ((size_t)(t * B_TOT) + b0 + m) * L_DIM + nb + n;
        float em_ = sigmf(a1[r] + bm);
        float es_ = softplusf(a2[r] + bs);
        sZ[m * L_DIM + nb + n] = f2bf(epsl[m * L_DIM + nb + n] * es_ + em_);
        em_out[gi] = em_; es_out[gi] = es_;
        float pm_ = pmv[r], ps_ = psv[r], d = em_ - pm_;
        kld_acc += 0.5f * (2.f * logf(ps_) - 2.f * logf(es_)
                           + (es_ * es_ + d * d) / (ps_ * ps_) - 1.f);
      }
    }
    __syncthreads();

    // S6: phi_z layer1 (K=64 from sZ) -> sT2
    for (int tl = wave; tl < 16; tl += 8)
      relu_store(sT2, 256, 0, tl, lane, pzb1,
                 wmma_tile(sZ, 64, sZ, 64, 64, wPZW1, 64, wPZW1, 64, 64, 64, tl * 16, lane, vz));
    __syncthreads();

    // S7: phi_z layer2 -> sAg[:,256:512]
    for (int tl = wave; tl < 16; tl += 8)
      relu_store(sAg, 1024, 256, tl, lane, pzb2,
                 wmma_tile(sT2, 256, sT2, 256, 256, wPZW2, 256, wPZW2, 256, 256, 256, tl * 16, lane, vz));
    __syncthreads();

    // S8: dec layer1 over [phi_z|h1] (split-A K=512) -> sT1
    for (int tl = wave; tl < 16; tl += 8)
      relu_store(sT1, 256, 0, tl, lane, db1,
                 wmma_tile(sAg + 256, 1024, sAg + 768, 1024, 256, wDW1, 512, wDW1, 512, 512, 512, tl * 16, lane, vz));
    __syncthreads();

    // S9: dec layer2 -> sT2
    for (int tl = wave; tl < 16; tl += 8)
      relu_store(sT2, 256, 0, tl, lane, db2,
                 wmma_tile(sT1, 256, sT1, 256, 256, wDW2, 256, wDW2, 256, 256, 256, tl * 16, lane, vz));
    __syncthreads();

    // S10: dec heads + NLL (dm tile `wave`, ds tile `wave`; 8 tiles each)
    {
      int nb = wave * 16;
      v8f a1 = wmma_tile(sT2, 256, sT2, 256, 256, wDMW, 256, wDMW, 256, 256, 256, nb, lane, vz);
      v8f a2 = wmma_tile(sT2, 256, sT2, 256, 256, wDSW, 256, wDSW, 256, 256, 256, nb, lane, vz);
      int n = lane & 15, mb = (lane >> 4) << 3;
      float bm = dmb[nb + n], bs = dsb[nb + n];
#pragma unroll
      for (int r = 0; r < 8; ++r) {
        int m = mb + r;
        size_t gi = ((size_t)(t * B_TOT) + b0 + m) * F_DIM + nb + n;
        float dmv = sigmf(a1[r] + bm);
        float dsv = softplusf(a2[r] + bs);
        dm_out[gi] = dmv; ds_out[gi] = dsv;
        float xv = xg[gi];
        nll_acc -= xv * logf(dmv) + (1.f - xv) * logf(1.f - dmv);
      }
    }
    __syncthreads();

    // S11: GRU layer 0.  A = sAg[:,0:768] = [gin|h0]; r/z fused K=768.
    {
      v8f hnew[2];
#pragma unroll
      for (int g = 0; g < 2; ++g) {
        int j = (wave * 2 + g) * 16;
        v8f ar  = wmma_tile(sAg, 1024, sAg, 1024, 768, wGIH0, 512, wGHH0, 256, 512, 768, j,        lane, vz);
        v8f az  = wmma_tile(sAg, 1024, sAg, 1024, 768, wGIH0, 512, wGHH0, 256, 512, 768, 256 + j,  lane, vz);
        v8f ain = wmma_tile(sAg, 1024, sAg, 1024, 512, wGIH0, 512, wGIH0, 512, 512, 512, 512 + j,  lane, vz);
        v8f ahn = wmma_tile(sAg + 512, 1024, sAg + 512, 1024, 256, wGHH0, 256, wGHH0, 256, 256, 256, 512 + j, lane, vz);
        int n = lane & 15, mb = (lane >> 4) << 3;
#pragma unroll
        for (int r = 0; r < 8; ++r) {
          float h0o = bf2f(sAg[(mb + r) * 1024 + 512 + j + n]);
          float rg = sigmf(ar[r]);
          float zg = sigmf(az[r]);
          float ng = tanhf(ain[r] + rg * ahn[r]);
          hnew[g][r] = (1.f - zg) * ng + zg * h0o;
        }
      }
      __syncthreads();
#pragma unroll
      for (int g = 0; g < 2; ++g) {
        int j = (wave * 2 + g) * 16, n = lane & 15, mb = (lane >> 4) << 3;
#pragma unroll
        for (int r = 0; r < 8; ++r)
          sAg[(mb + r) * 1024 + 512 + j + n] = f2bf(hnew[g][r]);
      }
      __syncthreads();
    }

    // S12: GRU layer 1.  A = sAg[:,512:1024] = [h0'|h1]; r/z fused K=512.
    {
      v8f hnew[2];
#pragma unroll
      for (int g = 0; g < 2; ++g) {
        int j = (wave * 2 + g) * 16;
        v8f ar  = wmma_tile(sAg + 512, 1024, sAg + 512, 1024, 512, wGIH1, 256, wGHH1, 256, 256, 512, j,       lane, vz);
        v8f az  = wmma_tile(sAg + 512, 1024, sAg + 512, 1024, 512, wGIH1, 256, wGHH1, 256, 256, 512, 256 + j, lane, vz);
        v8f ain = wmma_tile(sAg + 512, 1024, sAg + 512, 1024, 256, wGIH1, 256, wGIH1, 256, 256, 256, 512 + j, lane, vz);
        v8f ahn = wmma_tile(sAg + 768, 1024, sAg + 768, 1024, 256, wGHH1, 256, wGHH1, 256, 256, 256, 512 + j, lane, vz);
        int n = lane & 15, mb = (lane >> 4) << 3;
#pragma unroll
        for (int r = 0; r < 8; ++r) {
          float h1o = bf2f(sAg[(mb + r) * 1024 + 768 + j + n]);
          float rg = sigmf(ar[r]);
          float zg = sigmf(az[r]);
          float ng = tanhf(ain[r] + rg * ahn[r]);
          hnew[g][r] = (1.f - zg) * ng + zg * h1o;
        }
      }
      __syncthreads();
#pragma unroll
      for (int g = 0; g < 2; ++g) {
        int j = (wave * 2 + g) * 16, n = lane & 15, mb = (lane >> 4) << 3;
#pragma unroll
        for (int r = 0; r < 8; ++r)
          sAg[(mb + r) * 1024 + 768 + j + n] = f2bf(hnew[g][r]);
      }
      __syncthreads();
    }
  }

  // Final block reduction of kld/nll partials, atomic add into out[0],out[1]
  float* red = (float*)sT1;
  red[tid] = kld_acc;
  __syncthreads();
  for (int s = 128; s > 0; s >>= 1) { if (tid < s) red[tid] += red[tid + s]; __syncthreads(); }
  if (tid == 0) atomicAdd(&out[0], red[0]);
  __syncthreads();
  red[tid] = nll_acc;
  __syncthreads();
  for (int s = 128; s > 0; s >>= 1) { if (tid < s) red[tid] += red[tid + s]; __syncthreads(); }
  if (tid == 0) atomicAdd(&out[1], red[0]);
}

__global__ void cvt_bf16_kernel(const float* __restrict__ src, unsigned short* __restrict__ dst, int n) {
  int i = blockIdx.x * blockDim.x + threadIdx.x;
  if (i < n) dst[i] = f2bf(src[i]);
}

__global__ void init_out_kernel(float* out) { out[0] = 0.f; out[1] = 0.f; }

extern "C" void kernel_launch(void* const* d_in, const int* in_sizes, int n_in,
                              void* d_out, int out_size, void* d_ws, size_t ws_size,
                              hipStream_t stream) {
  unsigned short* wb = (unsigned short*)d_ws;
  auto cv = [&](int idx, size_t off, int n) {
    cvt_bf16_kernel<<<(n + 255) / 256, 256, 0, stream>>>((const float*)d_in[idx], wb + off, n);
  };
  cv(2,  OFF_PXW1, 256 * 128);
  cv(4,  OFF_PXW2, 256 * 256);
  cv(6,  OFF_PZW1, 256 * 64);
  cv(8,  OFF_PZW2, 256 * 256);
  cv(10, OFF_EW1,  256 * 512);
  cv(12, OFF_EW2,  256 * 256);
  cv(14, OFF_EMW,  64 * 256);
  cv(16, OFF_ESW,  64 * 256);
  cv(18, OFF_PRW,  256 * 256);
  cv(20, OFF_PMW,  64 * 256);
  cv(22, OFF_PSW,  64 * 256);
  cv(24, OFF_DW1,  256 * 512);
  cv(26, OFF_DW2,  256 * 256);
  cv(28, OFF_DMW,  128 * 256);
  cv(30, OFF_DSW,  128 * 256);
  cv(32, OFF_GIH0, 768 * 512);
  cv(33, OFF_GHH0, 768 * 256);
  cv(34, OFF_GIH1, 768 * 256);
  cv(35, OFF_GHH1, 768 * 256);

  init_out_kernel<<<1, 1, 0, stream>>>((float*)d_out);

  vrnn_persistent_kernel<<<dim3(B_TOT / 16), dim3(256), 0, stream>>>(
      (const float*)d_in[0], (const float*)d_in[1], (const unsigned short*)d_ws,
      (const float*)d_in[3],  (const float*)d_in[5],
      (const float*)d_in[7],  (const float*)d_in[9],
      (const float*)d_in[11], (const float*)d_in[13],
      (const float*)d_in[15], (const float*)d_in[17],
      (const float*)d_in[19], (const float*)d_in[21],
      (const float*)d_in[23], (const float*)d_in[25],
      (const float*)d_in[27], (const float*)d_in[29],
      (const float*)d_in[31], (float*)d_out);
}
